// scaled_dot_product_attention_46205258170906
// MI455X (gfx1250) — matrix operand order repaired and checked by reading, not compiled, not run on hardware
//
#include <hip/hip_runtime.h>

// ---------------------------------------------------------------------------
// Types for CDNA5 WMMA (wave32): bf16 16x16x32, f32 accumulate
// ---------------------------------------------------------------------------
typedef __bf16 v16bf __attribute__((ext_vector_type(16)));
typedef float v8f __attribute__((ext_vector_type(8)));
typedef unsigned short ushort8 __attribute__((ext_vector_type(8)));
typedef unsigned short ushort16 __attribute__((ext_vector_type(16)));

union FragU {
  ushort16 u;
  v16bf bf;
};

// Native f32 -> bf16 conversion (backend lowers to v_cvt*bf16* on gfx1250,
// replacing the 4-VALU-op manual RNE sequence).
__device__ __forceinline__ unsigned short f32_to_bf16(float f) {
  union { __bf16 b; unsigned short u; } v;
  v.b = (__bf16)f;
  return v.u;
}

__device__ __forceinline__ v16bf make_frag(ushort8 lo, ushort8 hi) {
  FragU f;
#pragma unroll
  for (int i = 0; i < 8; ++i) {
    f.u[i] = lo[i];
    f.u[i + 8] = hi[i];
  }
  return f.bf;
}

// A-matrix fragment (16x32, 16-bit): lane m = L%16, h = L>>4.
// Elements e<8 -> K = h*8+e ; e>=8 -> K = 16 + h*8 + (e-8).
__device__ __forceinline__ v16bf load_frag_a(const unsigned short* rowp, int h) {
  ushort8 lo = *(const ushort8*)(rowp + h * 8);
  ushort8 hi = *(const ushort8*)(rowp + 16 + h * 8);
  return make_frag(lo, hi);
}

// B-matrix fragment (32x16, 16-bit): lane n = L%16, h = L>>4.
// Elements e<8 -> K = h*8+e ; e>=8 -> K = 16 + h*8 + (e-8).
__device__ __forceinline__ v16bf load_frag_b(const unsigned short* rowp, int h) {
  ushort8 lo = *(const ushort8*)(rowp + h * 8);
  ushort8 hi = *(const ushort8*)(rowp + 16 + h * 8);
  return make_frag(lo, hi);
}

#define SEQ 2048
#define DEM 1024
#define NBATCH 4

// ---------------------------------------------------------------------------
// Kernel 1: f32 -> bf16 conversion
// ---------------------------------------------------------------------------
__global__ void cvt_kernel(const float* __restrict__ src,
                           unsigned short* __restrict__ dst, int n) {
  int i = blockIdx.x * blockDim.x + threadIdx.x;
  if (i < n) dst[i] = f32_to_bf16(src[i]);
}

// ---------------------------------------------------------------------------
// Kernel 2: fused QKV projection.  Y = Xb @ W^T, bf16 WMMA.
// One wave per block computes a 64x64 output tile: 4 A-frags x 4 B-frags
// -> 16 WMMAs per 32-wide k-step, 16 b128 loads per 16 WMMAs (32 FLOP/B).
// grid = (Srows/64, DEM/64, 3).
// ---------------------------------------------------------------------------
__global__ void __launch_bounds__(32) qkv_gemm(
    const unsigned short* __restrict__ xb,
    const unsigned short* __restrict__ wq,
    const unsigned short* __restrict__ wk,
    const unsigned short* __restrict__ wv,
    unsigned short* __restrict__ qb,
    unsigned short* __restrict__ kb,
    unsigned short* __restrict__ vb) {
  const int m0 = blockIdx.x * 64;
  const int n0 = blockIdx.y * 64;
  const unsigned short* W;
  unsigned short* O;
  if (blockIdx.z == 0) { W = wq; O = qb; }
  else if (blockIdx.z == 1) { W = wk; O = kb; }
  else { W = wv; O = vb; }

  const int lane = threadIdx.x & 31;
  const int c = lane & 15;   // A-row m / B-col n within a 16-tile
  const int h = lane >> 4;

  v8f acc[4][4];
#pragma unroll
  for (int mi = 0; mi < 4; ++mi)
#pragma unroll
    for (int ni = 0; ni < 4; ++ni)
#pragma unroll
      for (int r = 0; r < 8; ++r) acc[mi][ni][r] = 0.0f;

  const unsigned short* arow[4];
  const unsigned short* brow[4];
#pragma unroll
  for (int i = 0; i < 4; ++i) {
    arow[i] = xb + (size_t)(m0 + i * 16 + c) * DEM;
    brow[i] = W + (size_t)(n0 + i * 16 + c) * DEM;
  }

  for (int k0 = 0; k0 < DEM; k0 += 32) {
    v16bf a[4], b[4];
#pragma unroll
    for (int i = 0; i < 4; ++i) a[i] = load_frag_a(arow[i] + k0, h);
#pragma unroll
    for (int i = 0; i < 4; ++i) b[i] = load_frag_b(brow[i] + k0, h);
#pragma unroll
    for (int mi = 0; mi < 4; ++mi)
#pragma unroll
      for (int ni = 0; ni < 4; ++ni)
        acc[mi][ni] = __builtin_amdgcn_wmma_f32_16x16x32_bf16(
            false, a[mi], false, b[ni], (short)0, acc[mi][ni], false, false);
  }

  // C/D layout: lane holds column n=c, rows m = r + 8*h
#pragma unroll
  for (int mi = 0; mi < 4; ++mi)
#pragma unroll
    for (int ni = 0; ni < 4; ++ni)
#pragma unroll
      for (int r = 0; r < 8; ++r) {
        int m = m0 + mi * 16 + r + 8 * h;
        O[(size_t)m * DEM + n0 + ni * 16 + c] = f32_to_bf16(acc[mi][ni][r]);
      }
}

// ---------------------------------------------------------------------------
// Kernel 3: flash attention.  grid = (NBATCH * SEQ/16), block = 256 (8 waves).
// Each block: one 16-query tile, full 2048-key loop, online softmax.
// Wave w owns output columns [w*128, w*128+128).
// ---------------------------------------------------------------------------
__global__ void __launch_bounds__(256) attn_kernel(
    const unsigned short* __restrict__ Qb,
    const unsigned short* __restrict__ Kb,
    const unsigned short* __restrict__ Vb,
    float* __restrict__ out) {
  __shared__ __align__(16) unsigned short Qs[16 * DEM];  // 32 KB Q tile
  __shared__ __align__(16) float Ssc[16 * 128];          //  8 KB scores chunk
  __shared__ __align__(16) unsigned short Ps[16 * 128];  //  4 KB probs (bf16)
  __shared__ __align__(16) unsigned short Vt[DEM * 32];  // 64 KB V^T sub-chunk
  __shared__ float mrow[16], lrow[16], frow[16];

  const int tid = threadIdx.x;
  const int wave = tid >> 5;
  const int lane = tid & 31;
  const int c = lane & 15;  // A-row m / B-col n within tile
  const int h = lane >> 4;

  const int b = blockIdx.x >> 7;   // SEQ/16 = 128 q-tiles per batch
  const int qt = blockIdx.x & 127;
  const size_t qrow0 = (size_t)b * SEQ + (size_t)qt * 16;
  const unsigned short* Qbase = Qb + qrow0 * DEM;
  const unsigned short* Kbase = Kb + (size_t)b * SEQ * DEM;
  const unsigned short* Vbase = Vb + (size_t)b * SEQ * DEM;

  // Stage the 16x1024 Q tile into LDS with CDNA5 async global->LDS copies
  // (ASYNCcnt-tracked, no VGPR return data).  2048 x 16B packets, 8/thread,
  // uniform trip count so EXEC stays all-ones.
  {
    const unsigned lds_base = (unsigned)(unsigned long)(void*)&Qs[0];
#pragma unroll 1
    for (int i = tid; i < 16 * DEM / 8; i += 256) {
      unsigned lds_addr = lds_base + (unsigned)i * 16u;
      const unsigned short* g = Qbase + (size_t)i * 8;
      asm volatile("global_load_async_to_lds_b128 %0, %1, off"
                   :
                   : "v"(lds_addr), "v"(g)
                   : "memory");
    }
    asm volatile("s_wait_asynccnt 0" ::: "memory");
  }
  if (tid < 16) {
    mrow[tid] = -3.0e38f;
    lrow[tid] = 0.0f;
  }
  __syncthreads();

  v8f acc[8];
#pragma unroll
  for (int t = 0; t < 8; ++t)
#pragma unroll
    for (int r = 0; r < 8; ++r) acc[t][r] = 0.0f;

  const float scale = 0.03125f;  // 1/sqrt(1024)

  for (int kv0 = 0; kv0 < SEQ; kv0 += 128) {
    // ---- scores: wave w computes 16x16 tile at keys [kv0+16w, kv0+16w+16)
    {
      v8f s;
#pragma unroll
      for (int r = 0; r < 8; ++r) s[r] = 0.0f;
      const unsigned short* krow = Kbase + (size_t)(kv0 + wave * 16 + c) * DEM;
      const unsigned short* qrow = Qs + c * DEM;
      for (int d0 = 0; d0 < DEM; d0 += 32) {
        v16bf a = load_frag_a(qrow + d0, h);
        v16bf bf = load_frag_b(krow + d0, h);
        s = __builtin_amdgcn_wmma_f32_16x16x32_bf16(false, a, false, bf,
                                                    (short)0, s, false, false);
      }
#pragma unroll
      for (int r = 0; r < 8; ++r)
        Ssc[(r + 8 * h) * 128 + wave * 16 + c] = s[r] * scale;
    }
    __syncthreads();

    // ---- online softmax update over this 128-key chunk (one thread per row)
    if (tid < 16) {
      const int row = tid;
      float mc = -3.0e38f;
      for (int j = 0; j < 128; ++j) mc = fmaxf(mc, Ssc[row * 128 + j]);
      float mo = mrow[row];
      float mn = fmaxf(mo, mc);
      float fac = __expf(mo - mn);
      float sum = 0.0f;
      for (int j = 0; j < 128; ++j) {
        float p = __expf(Ssc[row * 128 + j] - mn);
        Ps[row * 128 + j] = f32_to_bf16(p);
        sum += p;
      }
      lrow[row] = lrow[row] * fac + sum;
      mrow[row] = mn;
      frow[row] = fac;
    }
    __syncthreads();

    // ---- rescale accumulators by exp(m_old - m_new) per row
    float fr[8];
#pragma unroll
    for (int r = 0; r < 8; ++r) fr[r] = frow[r + 8 * h];
#pragma unroll
    for (int t = 0; t < 8; ++t)
#pragma unroll
      for (int r = 0; r < 8; ++r) acc[t][r] *= fr[r];

    // ---- P @ V over 4 sub-chunks of 32 keys
    for (int sub = 0; sub < 4; ++sub) {
      __syncthreads();  // previous Vt consumers done
      {
        // stage V rows [kv0+32*sub, +32) transposed: Vt[d*32 + kv]
        const int kvr = tid & 31;
        const int dblk = tid >> 5;
        const unsigned short* vrow =
            Vbase + (size_t)(kv0 + sub * 32 + kvr) * DEM + dblk * 128;
#pragma unroll 1
        for (int d = 0; d < 128; d += 8) {
          ushort8 v = *(const ushort8*)(vrow + d);
#pragma unroll
          for (int e = 0; e < 8; ++e)
            Vt[(dblk * 128 + d + e) * 32 + kvr] = v[e];
        }
      }
      __syncthreads();
      const unsigned short* prow = Ps + c * 128 + sub * 32;
#pragma unroll
      for (int t = 0; t < 8; ++t) {
        v16bf a = load_frag_a(prow, h);
        const int dv = wave * 128 + t * 16 + c;
        v16bf bf = load_frag_b(Vt + dv * 32, h);
        acc[t] = __builtin_amdgcn_wmma_f32_16x16x32_bf16(
            false, a, false, bf, (short)0, acc[t], false, false);
      }
    }
    __syncthreads();
  }

  // ---- epilogue: normalize by 1/l and store f32 output
  float li[8];
#pragma unroll
  for (int r = 0; r < 8; ++r) li[r] = 1.0f / lrow[r + 8 * h];
  float* obase = out + qrow0 * DEM;
#pragma unroll
  for (int t = 0; t < 8; ++t) {
    const int dv = wave * 128 + t * 16 + c;
#pragma unroll
    for (int r = 0; r < 8; ++r)
      obase[(size_t)(r + 8 * h) * DEM + dv] = acc[t][r] * li[r];
  }
}

// ---------------------------------------------------------------------------
// Launch
// ---------------------------------------------------------------------------
extern "C" void kernel_launch(void* const* d_in, const int* in_sizes, int n_in,
                              void* d_out, int out_size, void* d_ws,
                              size_t ws_size, hipStream_t stream) {
  const float* x = (const float*)d_in[0];
  const float* Wq = (const float*)d_in[1];
  const float* Wk = (const float*)d_in[2];
  const float* Wv = (const float*)d_in[3];
  float* out = (float*)d_out;

  const size_t NX = (size_t)NBATCH * SEQ * DEM;  // 8388608
  const size_t NW = (size_t)DEM * DEM;           // 1048576

  unsigned short* ws = (unsigned short*)d_ws;
  unsigned short* xb = ws;
  unsigned short* wqb = xb + NX;
  unsigned short* wkb = wqb + NW;
  unsigned short* wvb = wkb + NW;
  unsigned short* qb = wvb + NW;
  unsigned short* kb = qb + NX;
  unsigned short* vb = kb + NX;

  cvt_kernel<<<(int)((NX + 255) / 256), 256, 0, stream>>>(x, xb, (int)NX);
  cvt_kernel<<<(int)((NW + 255) / 256), 256, 0, stream>>>(Wq, wqb, (int)NW);
  cvt_kernel<<<(int)((NW + 255) / 256), 256, 0, stream>>>(Wk, wkb, (int)NW);
  cvt_kernel<<<(int)((NW + 255) / 256), 256, 0, stream>>>(Wv, wvb, (int)NW);

  qkv_gemm<<<dim3(NBATCH * SEQ / 64, DEM / 64, 3), 32, 0, stream>>>(
      xb, wqb, wkb, wvb, qb, kb, vb);

  attn_kernel<<<NBATCH * SEQ / 16, 256, 0, stream>>>(qb, kb, vb, out);
}
